// TinyLLM_17085379903951
// MI455X (gfx1250) — matrix operand order repaired and checked by reading, not compiled, not run on hardware
//
#include <hip/hip_runtime.h>
#include <hip/hip_bf16.h>
#include <math.h>

// ---------------------------------------------------------------------------
// GPT-2-small forward on MI455X (gfx1250, wave32, WMMA bf16 16x16x32).
// All GEMMs bf16 with f32 accumulation; LN/softmax fp32.
// Weights are converted AND transposed to [N,K] bf16 once per launch, so every
// GEMM stages A and B tiles as contiguous-K b128 copies (async global->LDS,
// builtin confirmed present on this toolchain). V is written by the QKV GEMM
// epilogue in transposed [B,H,dk,T] layout so flash attention needs no V
// staging and no block barriers.
// ---------------------------------------------------------------------------

typedef __attribute__((ext_vector_type(16))) __bf16 v16bf;
typedef __attribute__((ext_vector_type(8)))  float  v8f;
typedef int v4i_gcc __attribute__((vector_size(16)));   // matches builtin param

union FragBF { v16bf v; uint4 q[2]; };

#define LLM_B  2
#define LLM_T  1024
#define LLM_D  768
#define LLM_H  12
#define LLM_DK 64
#define LLM_FF 3072
#define LLM_L  12
#define LLM_V  50257

static_assert(sizeof(__hip_bfloat16) == 2, "bf16 size");

// ---------------- async global -> LDS copy (CDNA5), with safe fallback ------
#if defined(__has_builtin)
#if __has_builtin(__builtin_amdgcn_global_load_async_to_lds_b128)
#define HAVE_ASYNC_LDS 1
#endif
#endif
#ifndef HAVE_ASYNC_LDS
#define HAVE_ASYNC_LDS 0
#endif

__device__ __forceinline__ void cp16_g2l(__hip_bfloat16* dst_lds,
                                         const __hip_bfloat16* src) {
#if HAVE_ASYNC_LDS
  __builtin_amdgcn_global_load_async_to_lds_b128(
      (__attribute__((address_space(1))) v4i_gcc*)(src),
      (__attribute__((address_space(3))) v4i_gcc*)(dst_lds), 0, 0);
#else
  *(uint4*)dst_lds = *(const uint4*)src;
#endif
}

__device__ __forceinline__ void wait_async_lds() {
#if HAVE_ASYNC_LDS
#if __has_builtin(__builtin_amdgcn_s_wait_asynccnt)
  __builtin_amdgcn_s_wait_asynccnt(0);
#else
  asm volatile("s_wait_asynccnt 0" ::: "memory");
#endif
#endif
}

// ---------------- ds_swizzle XOR reductions (no address setup) --------------
template <int MASK>
__device__ __forceinline__ float swz_f(float v) {
  return __builtin_bit_cast(
      float, __builtin_amdgcn_ds_swizzle(__builtin_bit_cast(int, v),
                                         0x1f | (MASK << 10)));
}
__device__ __forceinline__ float half_max16(float v) {   // within 16-lane halves
  v = fmaxf(v, swz_f<1>(v)); v = fmaxf(v, swz_f<2>(v));
  v = fmaxf(v, swz_f<4>(v)); v = fmaxf(v, swz_f<8>(v));
  return v;
}
__device__ __forceinline__ float half_sum16(float v) {
  v += swz_f<1>(v); v += swz_f<2>(v); v += swz_f<4>(v); v += swz_f<8>(v);
  return v;
}
__device__ __forceinline__ float wave_sum32(float v) {
  v = half_sum16(v); v += swz_f<16>(v); return v;
}

// ---------------------------------------------------------------------------
// fp32 -> bf16 bulk convert (no transpose; tok_emb is already [V, K=D])
// ---------------------------------------------------------------------------
__global__ void cvt_bf16_kernel(const float* __restrict__ src,
                                __hip_bfloat16* __restrict__ dst, long n) {
  long i = (long)blockIdx.x * blockDim.x + threadIdx.x;
  long stride = (long)gridDim.x * blockDim.x;
  for (; i < n; i += stride) dst[i] = __float2bfloat16(src[i]);
}

// fp32 [K,N] -> bf16 [N,K] per layer (blockIdx.y = layer, blockIdx.x = k row)
__global__ void cvtT_kernel(const float* __restrict__ src,
                            __hip_bfloat16* __restrict__ dst, int K, int N) {
  long base = (long)blockIdx.y * K * N;
  int k = blockIdx.x;
  const float* s = src + base + (long)k * N;
  __hip_bfloat16* d = dst + base;
  for (int n = threadIdx.x; n < N; n += blockDim.x)
    d[(long)n * K + k] = __float2bfloat16(s[n]);
}

// ---------------------------------------------------------------------------
// token embedding + sinusoidal PE (fp32 residual stream)
// ---------------------------------------------------------------------------
__global__ void embed_kernel(const int* __restrict__ idx,
                             const float* __restrict__ emb,
                             float* __restrict__ x) {
  int row = blockIdx.x;                 // 0 .. B*T-1
  int t = row % LLM_T;
  const float* e = emb + (long)idx[row] * LLM_D;
  float* xo = x + (long)row * LLM_D;
  const float c0 = -9.21034037f / (float)LLM_D;   // -ln(10000)/D
  for (int d = threadIdx.x; d < LLM_D; d += blockDim.x) {
    int i2 = d & ~1;
    float ang = (float)t * __expf(c0 * (float)i2);
    xo[d] = e[d] + ((d & 1) ? __cosf(ang) : __sinf(ang));
  }
}

// ---------------------------------------------------------------------------
// LayerNorm fp32 -> bf16, one 256-thread block per row
// ---------------------------------------------------------------------------
__global__ void layernorm_kernel(const float* __restrict__ x,
                                 const float* __restrict__ w,
                                 const float* __restrict__ b,
                                 __hip_bfloat16* __restrict__ out) {
  __shared__ float smem[16];
  int row = blockIdx.x;
  const float* xr = x + (long)row * LLM_D;
  float s = 0.f, s2 = 0.f;
  for (int d = threadIdx.x; d < LLM_D; d += blockDim.x) {
    float v = xr[d]; s += v; s2 += v * v;
  }
  s = wave_sum32(s); s2 = wave_sum32(s2);
  int wid = threadIdx.x >> 5, lid = threadIdx.x & 31;
  if (lid == 0) { smem[wid] = s; smem[8 + wid] = s2; }
  __syncthreads();
  if (threadIdx.x == 0) {
    float a = 0.f, a2 = 0.f;
    int nw = blockDim.x >> 5;
    for (int i = 0; i < nw; i++) { a += smem[i]; a2 += smem[8 + i]; }
    smem[0] = a; smem[8] = a2;
  }
  __syncthreads();
  float mu = smem[0] / LLM_D;
  float var = smem[8] / LLM_D - mu * mu;
  float rstd = rsqrtf(var + 1e-5f);
  __hip_bfloat16* o = out + (long)row * LLM_D;
  for (int d = threadIdx.x; d < LLM_D; d += blockDim.x)
    o[d] = __float2bfloat16((xr[d] - mu) * rstd * w[d] + b[d]);
}

// ---------------------------------------------------------------------------
// bf16 WMMA GEMM: C[M,N] = A[M,K] @ B^T, with B stored [N,K].
// 128 threads = 4 waves (2x2); block tile 64x64, K-step 32; each wave 2x2
// v_wmma_f32_16x16x32_bf16. A and B tiles staged via async b128 g->LDS.
// EPI: 0 qkv split store (q,k -> [M,2D], v -> vT[B,H,dk,T])
//      1 bias + exact GELU -> bf16 | 2 resid+bias -> f32
//      3 resid -> f32             | 4 f32 store (ragged N guarded)
// ---------------------------------------------------------------------------
template <int EPI>
__global__ __launch_bounds__(128)
void gemm_bf16_kernel(const __hip_bfloat16* __restrict__ A,
                      const __hip_bfloat16* __restrict__ Bm,   // [N,K]
                      const float* __restrict__ bias,
                      const float* __restrict__ resid,
                      float* __restrict__ outF,
                      __hip_bfloat16* __restrict__ outB,
                      __hip_bfloat16* __restrict__ outV,       // vT (EPI==0)
                      int M, int N, int K) {
  const int BKP = 40;                      // padded K stride (80B)
  __shared__ __hip_bfloat16 As[64 * BKP];
  __shared__ __hip_bfloat16 Bs[64 * BKP];
  int bn0 = blockIdx.x * 64, bm0 = blockIdx.y * 64;
  int tid = threadIdx.x, lane = tid & 31, w = tid >> 5;
  int wm = (w >> 1) * 32, wn = (w & 1) * 32;
  v8f acc[2][2] = {};

  for (int k0 = 0; k0 < K; k0 += 32) {
    __syncthreads();                       // previous tile fully consumed
#pragma unroll
    for (int it = 0; it < 2; it++) {
      int c = tid + it * 128;
      int r = c >> 2, kc = (c & 3) * 8;    // r = row of A tile / row of B tile
      const __hip_bfloat16* ap = A + (long)(bm0 + r) * K + k0 + kc;
      if (k0 + 32 < K) __builtin_prefetch(ap + 32, 0, 1);
      cp16_g2l(&As[r * BKP + kc], ap);
      if (bn0 + r < N) {
        cp16_g2l(&Bs[r * BKP + kc], Bm + (long)(bn0 + r) * K + k0 + kc);
      } else {
        uint4 z = {0u, 0u, 0u, 0u};
        *(uint4*)(&Bs[r * BKP + kc]) = z;
      }
    }
    wait_async_lds();
    __syncthreads();

    FragBF a[2], bf[2];
    int akb = (lane >> 4) * 8;
#pragma unroll
    for (int mi = 0; mi < 2; mi++) {
      int m = wm + mi * 16 + (lane & 15);
      a[mi].q[0] = *(const uint4*)(&As[m * BKP + akb]);
      a[mi].q[1] = *(const uint4*)(&As[m * BKP + 16 + akb]);
    }
    int bkb = (lane >> 4) * 8;
#pragma unroll
    for (int ni = 0; ni < 2; ni++) {
      int n = wn + ni * 16 + (lane & 15);
      bf[ni].q[0] = *(const uint4*)(&Bs[n * BKP + bkb]);
      bf[ni].q[1] = *(const uint4*)(&Bs[n * BKP + bkb + 16]);
    }
#pragma unroll
    for (int mi = 0; mi < 2; mi++)
#pragma unroll
      for (int ni = 0; ni < 2; ni++)
        acc[mi][ni] = __builtin_amdgcn_wmma_f32_16x16x32_bf16(
            false, a[mi].v, false, bf[ni].v, (short)0, acc[mi][ni], false, false);
  }

  int hf = lane >> 4, nl = lane & 15;
#pragma unroll
  for (int mi = 0; mi < 2; mi++)
#pragma unroll
    for (int ni = 0; ni < 2; ni++) {
      int col = bn0 + wn + ni * 16 + nl;
      if (col >= N) continue;
#pragma unroll
      for (int r = 0; r < 8; r++) {
        int row = bm0 + wm + mi * 16 + r + hf * 8;
        float v = acc[mi][ni][r];
        if (EPI == 0) {
          if (col < 2 * LLM_D) {           // q,k packed [M, 2D]
            outB[(long)row * 2 * LLM_D + col] = __float2bfloat16(v);
          } else {                          // v -> vT[B,H,dk,T]
            int c = col - 2 * LLM_D;
            int hh = c >> 6, dl = c & (LLM_DK - 1);
            int bq = row >> 10, t = row & (LLM_T - 1);
            outV[(((long)bq * LLM_H + hh) * LLM_DK + dl) * LLM_T + t] =
                __float2bfloat16(v);
          }
        } else if (EPI == 1) {
          float tt = v + bias[col];
          float g = 0.5f * tt * (1.f + erff(tt * 0.70710678f));
          outB[(long)row * N + col] = __float2bfloat16(g);
        } else if (EPI == 2) {
          long o = (long)row * N + col;
          outF[o] = resid[o] + v + bias[col];
        } else if (EPI == 3) {
          long o = (long)row * N + col;
          outF[o] = resid[o] + v;
        } else {
          outF[(long)row * N + col] = v;
        }
      }
    }
}

// ---------------------------------------------------------------------------
// Flash causal attention, barrier-free. Block = (64-query strip, head, batch),
// 4 waves x 16 query rows, each wave loops only to its own causal bound.
// k^T fragments contiguous in qk buffer; V fragments contiguous in vT.
// Only the per-wave probs C-layout -> A-layout round-trip touches LDS
// (same-wave LDS is in-order; compiler barrier stops reordering).
// ---------------------------------------------------------------------------
__global__ __launch_bounds__(128)
void attn_kernel(const __hip_bfloat16* __restrict__ qk,   // [B*T, 2D]
                 const __hip_bfloat16* __restrict__ vT,   // [B,H,dk,T]
                 __hip_bfloat16* __restrict__ obuf) {
  const int BKP = 40;
  __shared__ __hip_bfloat16 Pp[4][16 * BKP];
  int tile = blockIdx.x, h = blockIdx.y, bb = blockIdx.z;
  int tid = threadIdx.x, lane = tid & 31, w = tid >> 5;
  int i0 = tile * 64 + w * 16;
  int hf = lane >> 4, nl = lane & 15;
  const long rs = 2 * LLM_D;
  const __hip_bfloat16* base = qk + (long)bb * LLM_T * rs;
  const __hip_bfloat16* vb0 = vT + ((long)bb * LLM_H + h) * LLM_DK * LLM_T;

  FragBF qa[2];
  {
    int qi = i0 + nl;
    int akb = hf * 8;
    const __hip_bfloat16* qp = base + (long)qi * rs + h * LLM_DK;
#pragma unroll
    for (int ks = 0; ks < 2; ks++) {
      qa[ks].q[0] = *(const uint4*)(qp + ks * 32 + akb);
      qa[ks].q[1] = *(const uint4*)(qp + ks * 32 + 16 + akb);
    }
  }

  v8f o[4] = {};
  float mprev[8], lsum[8];
#pragma unroll
  for (int r = 0; r < 8; r++) { mprev[r] = -1e30f; lsum[r] = 0.f; }

  int jend = i0 + 16;                      // wave-local causal bound
  for (int j0 = 0; j0 < jend; j0 += 32) {
    // ---- scores: q @ k^T
    v8f s[2] = {};
#pragma unroll
    for (int nt = 0; nt < 2; nt++) {
      int j = j0 + nt * 16 + nl;
      int bkb = hf * 8;
      const __hip_bfloat16* kp = base + (long)j * rs + LLM_D + h * LLM_DK;
      FragBF k0f, k1f;
      k0f.q[0] = *(const uint4*)(kp + bkb);
      k0f.q[1] = *(const uint4*)(kp + bkb + 16);
      k1f.q[0] = *(const uint4*)(kp + 32 + bkb);
      k1f.q[1] = *(const uint4*)(kp + 32 + bkb + 16);
      s[nt] = __builtin_amdgcn_wmma_f32_16x16x32_bf16(
          false, qa[0].v, false, k0f.v, (short)0, s[nt], false, false);
      s[nt] = __builtin_amdgcn_wmma_f32_16x16x32_bf16(
          false, qa[1].v, false, k1f.v, (short)0, s[nt], false, false);
    }
    // ---- causal mask + online softmax (rows live in 16-lane halves)
#pragma unroll
    for (int r = 0; r < 8; r++) {
      int qi = i0 + r + hf * 8;
      float s0 = s[0][r] * 0.125f;          // 1/sqrt(64)
      float s1 = s[1][r] * 0.125f;
      if (j0 + nl > qi)      s0 = -1e30f;
      if (j0 + 16 + nl > qi) s1 = -1e30f;
      float mx = half_max16(fmaxf(s0, s1));
      float newm = fmaxf(mprev[r], mx);
      float p0 = __expf(s0 - newm), p1 = __expf(s1 - newm);
      float ls = half_sum16(p0 + p1);
      float fac = __expf(mprev[r] - newm);
      lsum[r] = lsum[r] * fac + ls;
#pragma unroll
      for (int nt = 0; nt < 4; nt++) o[nt][r] *= fac;
      mprev[r] = newm;
      int m = r + hf * 8;
      Pp[w][m * BKP + nl]      = __float2bfloat16(p0);
      Pp[w][m * BKP + 16 + nl] = __float2bfloat16(p1);
    }
    asm volatile("" ::: "memory");          // same-wave LDS is in-order
    // ---- probs @ V (V fragments contiguous in vT)
    FragBF pa;
    {
      int m = nl, akb = hf * 8;
      pa.q[0] = *(const uint4*)(&Pp[w][m * BKP + akb]);
      pa.q[1] = *(const uint4*)(&Pp[w][m * BKP + 16 + akb]);
    }
    int bkb = hf * 8;
#pragma unroll
    for (int nt = 0; nt < 4; nt++) {
      int n = nt * 16 + nl;                 // d index
      const __hip_bfloat16* vp = vb0 + (long)n * LLM_T + j0 + bkb;
      FragBF vbf;
      vbf.q[0] = *(const uint4*)(vp);
      vbf.q[1] = *(const uint4*)(vp + 16);
      o[nt] = __builtin_amdgcn_wmma_f32_16x16x32_bf16(
          false, pa.v, false, vbf.v, (short)0, o[nt], false, false);
    }
  }

  // ---- normalize + store o (concat heads) as bf16
  __hip_bfloat16* op = obuf + (long)bb * LLM_T * LLM_D;
#pragma unroll
  for (int r = 0; r < 8; r++) {
    float inv = 1.f / lsum[r];
    int qi = i0 + r + hf * 8;
#pragma unroll
    for (int nt = 0; nt < 4; nt++)
      op[(long)qi * LLM_D + h * LLM_DK + nt * 16 + nl] =
          __float2bfloat16(o[nt][r] * inv);
  }
}

// ---------------------------------------------------------------------------
// Host orchestration (all launches on `stream`; graph-capture safe)
// ---------------------------------------------------------------------------
extern "C" void kernel_launch(void* const* d_in, const int* in_sizes, int n_in,
                              void* d_out, int out_size, void* d_ws, size_t ws_size,
                              hipStream_t stream) {
  (void)in_sizes; (void)n_in; (void)out_size; (void)ws_size;
  const int*   idx    = (const int*)  d_in[0];
  const float* tokemb = (const float*)d_in[1];
  const float* ln1w   = (const float*)d_in[2];
  const float* ln1b   = (const float*)d_in[3];
  const float* qkvw   = (const float*)d_in[4];
  const float* outw   = (const float*)d_in[5];
  const float* ln2w   = (const float*)d_in[6];
  const float* ln2b   = (const float*)d_in[7];
  const float* ffw1   = (const float*)d_in[8];
  const float* ffb1   = (const float*)d_in[9];
  const float* ffw2   = (const float*)d_in[10];
  const float* ffb2   = (const float*)d_in[11];
  const float* lnfw   = (const float*)d_in[12];
  const float* lnfb   = (const float*)d_in[13];

  const long M = (long)LLM_B * LLM_T;           // 2048 rows
  const long nqkvw = (long)LLM_L * LLM_D * 3 * LLM_D;
  const long noutw = (long)LLM_L * LLM_D * LLM_D;
  const long nff1w = (long)LLM_L * LLM_D * LLM_FF;
  const long nff2w = (long)LLM_L * LLM_FF * LLM_D;
  const long nemb  = (long)LLM_V * LLM_D;

  char* ws = (char*)d_ws;
  auto alloc = [&](size_t bytes) -> char* {
    char* p = ws; ws += (bytes + 255) & ~(size_t)255; return p;
  };
  // transposed bf16 weights [N,K] per layer
  __hip_bfloat16* qkvw_t = (__hip_bfloat16*)alloc(nqkvw * 2);
  __hip_bfloat16* outw_t = (__hip_bfloat16*)alloc(noutw * 2);
  __hip_bfloat16* ff1w_t = (__hip_bfloat16*)alloc(nff1w * 2);
  __hip_bfloat16* ff2w_t = (__hip_bfloat16*)alloc(nff2w * 2);
  __hip_bfloat16* emb_bf = (__hip_bfloat16*)alloc(nemb * 2);   // already [V,K]
  float*          x      = (float*)         alloc(M * LLM_D * 4);
  __hip_bfloat16* hbuf   = (__hip_bfloat16*)alloc(M * LLM_D * 2);
  __hip_bfloat16* qkact  = (__hip_bfloat16*)alloc(M * 2 * LLM_D * 2);
  __hip_bfloat16* vTact  = (__hip_bfloat16*)alloc((long)LLM_B * LLM_H * LLM_DK * LLM_T * 2);
  __hip_bfloat16* obuf   = (__hip_bfloat16*)alloc(M * LLM_D * 2);
  __hip_bfloat16* ffact  = (__hip_bfloat16*)alloc(M * LLM_FF * 2);

  // weight convert (+transpose) to bf16, once per launch
  cvtT_kernel<<<dim3(LLM_D,  LLM_L), 256, 0, stream>>>(qkvw, qkvw_t, LLM_D, 3 * LLM_D);
  cvtT_kernel<<<dim3(LLM_D,  LLM_L), 256, 0, stream>>>(outw, outw_t, LLM_D, LLM_D);
  cvtT_kernel<<<dim3(LLM_D,  LLM_L), 256, 0, stream>>>(ffw1, ff1w_t, LLM_D, LLM_FF);
  cvtT_kernel<<<dim3(LLM_FF, LLM_L), 256, 0, stream>>>(ffw2, ff2w_t, LLM_FF, LLM_D);
  cvt_bf16_kernel<<<4096, 256, 0, stream>>>(tokemb, emb_bf, nemb);

  embed_kernel<<<(int)M, 256, 0, stream>>>(idx, tokemb, x);

  dim3 blk(128);
  const int MT = (int)(M / 64);                 // 32 row tiles
  for (int l = 0; l < LLM_L; l++) {
    layernorm_kernel<<<(int)M, 256, 0, stream>>>(x, ln1w + l * LLM_D, ln1b + l * LLM_D, hbuf);
    // qkv = h @ qkv_w : q,k -> qkact, v -> vTact (transposed)
    gemm_bf16_kernel<0><<<dim3(3 * LLM_D / 64, MT), blk, 0, stream>>>(
        hbuf, qkvw_t + (long)l * LLM_D * 3 * LLM_D,
        nullptr, nullptr, nullptr, qkact, vTact, (int)M, 3 * LLM_D, LLM_D);
    attn_kernel<<<dim3(LLM_T / 64, LLM_H, LLM_B), blk, 0, stream>>>(qkact, vTact, obuf);
    // x = x + o @ out_w
    gemm_bf16_kernel<3><<<dim3(LLM_D / 64, MT), blk, 0, stream>>>(
        obuf, outw_t + (long)l * LLM_D * LLM_D,
        nullptr, x, x, nullptr, nullptr, (int)M, LLM_D, LLM_D);
    layernorm_kernel<<<(int)M, 256, 0, stream>>>(x, ln2w + l * LLM_D, ln2b + l * LLM_D, hbuf);
    // f = gelu(h @ w1 + b1)
    gemm_bf16_kernel<1><<<dim3(LLM_FF / 64, MT), blk, 0, stream>>>(
        hbuf, ff1w_t + (long)l * LLM_D * LLM_FF,
        ffb1 + (long)l * LLM_FF, nullptr, nullptr, ffact, nullptr,
        (int)M, LLM_FF, LLM_D);
    // x = x + f @ w2 + b2
    gemm_bf16_kernel<2><<<dim3(LLM_D / 64, MT), blk, 0, stream>>>(
        ffact, ff2w_t + (long)l * LLM_FF * LLM_D,
        ffb2 + (long)l * LLM_D, x, x, nullptr, nullptr, (int)M, LLM_D, LLM_FF);
  }
  // final LN + tied lm_head (tok_emb is already [V, D] = [N, K])
  layernorm_kernel<<<(int)M, 256, 0, stream>>>(x, lnfw, lnfb, hbuf);
  gemm_bf16_kernel<4><<<dim3((LLM_V + 63) / 64, MT), blk, 0, stream>>>(
      hbuf, emb_bf, nullptr, nullptr, (float*)d_out, nullptr, nullptr,
      (int)M, LLM_V, LLM_D);
}
